// AtlasMemoryLayer_35055523070483
// MI455X (gfx1250) — matrix operand order repaired and checked by reading, not compiled, not run on hardware
//
#include <hip/hip_runtime.h>
#include <hip/hip_bf16.h>
#include <stdint.h>

#define B_ 2
#define T_ 1024
#define C_ 1024
#define H_ 8
#define D_ 64
#define DI_ 512
#define CS_ 64
#define NC_ 16
#define OMEGA_ 16
#define BT_ (B_ * T_)

typedef __attribute__((ext_vector_type(16))) __bf16 v16bf;
typedef __attribute__((ext_vector_type(8)))  __bf16 v8bf;
typedef __attribute__((ext_vector_type(8)))  float  v8f;
typedef __attribute__((ext_vector_type(4)))  uint32_t u32x4;
typedef __attribute__((ext_vector_type(8)))  int      i32x8;
typedef __attribute__((ext_vector_type(4)))  int      i32x4;

__device__ __forceinline__ __bf16 f2bf(float f) {
  union { float f; uint32_t u; } c; c.f = f;
  uint32_t u = c.u + 0x7FFFu + ((c.u >> 16) & 1u);   // round-to-nearest-even
  uint16_t h = (uint16_t)(u >> 16);
  __bf16 r; __builtin_memcpy(&r, &h, 2);
  return r;
}

__device__ __forceinline__ v16bf cat8(v8bf lo, v8bf hi) {
  return __builtin_shufflevector(lo, hi, 0, 1, 2, 3, 4, 5, 6, 7,
                                 8, 9, 10, 11, 12, 13, 14, 15);
}

__device__ __forceinline__ float sigmoidf(float z) {
  return 1.0f / (1.0f + __expf(-z));
}

// ---------------------------------------------------------------------------
// One-shot f32 -> bf16 conversion (keeps all WMMA inner loops conversion-free)
// ---------------------------------------------------------------------------
__global__ void cvt_bf16_kernel(const float* __restrict__ in,
                                __bf16* __restrict__ out, int n) {
  const int i = blockIdx.x * 256 + threadIdx.x;
  if (i < n) out[i] = f2bf(in[i]);
}

// ---------------------------------------------------------------------------
// Y[M][N] = X[M][K] @ W[N][K]^T, X/W already bf16 row-major (K contiguous).
// bf16 WMMA 16x16x32, f32 accumulate. One wave -> 16x64 output strip.
// ---------------------------------------------------------------------------
__global__ void gemm_bf16_kernel(const __bf16* __restrict__ X,
                                 const __bf16* __restrict__ W,
                                 float* __restrict__ Y,
                                 int M, int N, int Kdim) {
  const int lane = threadIdx.x & 31;
  const int wave = threadIdx.x >> 5;
  const int gw   = blockIdx.x * 8 + wave;
  const int ntiles = N >> 6;
  const int mrow = gw / ntiles;
  const int ncol = (gw % ntiles) << 6;
  if (mrow * 16 >= M) return;

  const __bf16* arow = X + (size_t)(mrow * 16 + (lane & 15)) * Kdim;
  v8f acc[4] = {};
  const int offA = (lane < 16) ? 0 : 8;
  const int offB = (lane < 16) ? 0 : 8;

  for (int kb = 0; kb < Kdim; kb += 32) {
    __builtin_prefetch(arow + kb + 256, 0, 1);
    const v16bf a = cat8(*(const v8bf*)(arow + kb + offA),
                         *(const v8bf*)(arow + kb + 16 + offA));
#pragma unroll
    for (int t = 0; t < 4; ++t) {
      const __bf16* brow = W + (size_t)(ncol + t * 16 + (lane & 15)) * Kdim;
      const v16bf bv = cat8(*(const v8bf*)(brow + kb + offB),
                            *(const v8bf*)(brow + kb + 16 + offB));
      acc[t] = __builtin_amdgcn_wmma_f32_16x16x32_bf16(
          false, a, false, bv, (short)0, acc[t], false, false);
    }
  }
  const int rb = (lane < 16) ? 0 : 8;
  const int nc = lane & 15;
#pragma unroll
  for (int t = 0; t < 4; ++t)
#pragma unroll
    for (int r = 0; r < 8; ++r)
      Y[(size_t)(mrow * 16 + rb + r) * N + ncol + t * 16 + nc] = acc[t][r];
}

// ---------------------------------------------------------------------------
// Gate projections: alpha/eta/theta/gamma/rg = sigmoid(x@w^T + b), eta*=0.1
// ---------------------------------------------------------------------------
__global__ void gates_kernel(const float* __restrict__ x,
                             const float* __restrict__ ga_w, const float* __restrict__ ga_b,
                             const float* __restrict__ ge_w, const float* __restrict__ ge_b,
                             const float* __restrict__ gt_w, const float* __restrict__ gt_b,
                             const float* __restrict__ gg_w, const float* __restrict__ gg_b,
                             const float* __restrict__ rg_w,
                             float* __restrict__ alpha, float* __restrict__ eta,
                             float* __restrict__ theta, float* __restrict__ gammag,
                             float* __restrict__ rgate) {
  const int bt = blockIdx.x;
  __shared__ float sx[C_];
  __shared__ float sacc[40];
  for (int i = threadIdx.x; i < C_; i += 256) sx[i] = x[(size_t)bt * C_ + i];
  if (threadIdx.x < 40) sacc[threadIdx.x] = 0.0f;
  __syncthreads();
  for (int o = 0; o < 40; ++o) {
    const int g = o >> 3, h = o & 7;
    const float* w = (g == 0 ? ga_w : g == 1 ? ge_w : g == 2 ? gt_w : g == 3 ? gg_w : rg_w)
                     + (size_t)h * C_;
    float p = 0.0f;
    for (int k = threadIdx.x; k < C_; k += 256) p += sx[k] * w[k];
    atomicAdd(&sacc[o], p);
  }
  __syncthreads();
  if (threadIdx.x < 40) {
    const int g = threadIdx.x >> 3, h = threadIdx.x & 7;
    float bias = (g == 0 ? ga_b[h] : g == 1 ? ge_b[h] : g == 2 ? gt_b[h] : g == 3 ? gg_b[h] : 0.0f);
    float sv = sigmoidf(sacc[threadIdx.x] + bias);
    if (g == 1) sv *= 0.1f;                           // MAX_LR
    float* outp = (g == 0 ? alpha : g == 1 ? eta : g == 2 ? theta : g == 3 ? gammag : rgate);
    outp[(size_t)bt * H_ + h] = sv;
  }
}

// ---------------------------------------------------------------------------
// Causal depthwise conv (K=4) + bias + polynomial feature map on k.
// ---------------------------------------------------------------------------
__global__ void conv_poly_kernel(const float* __restrict__ qraw,
                                 const float* __restrict__ kraw,
                                 const float* __restrict__ vraw,
                                 const float* __restrict__ cqw, const float* __restrict__ cqb,
                                 const float* __restrict__ ckw, const float* __restrict__ ckb,
                                 const float* __restrict__ cvw, const float* __restrict__ cvb,
                                 const float* __restrict__ poly,
                                 float* __restrict__ qc, float* __restrict__ kphi,
                                 float* __restrict__ vc) {
  const size_t idx = (size_t)blockIdx.x * 256 + threadIdx.x;
  if (idx >= (size_t)BT_ * DI_) return;
  const int d = (int)(idx % DI_);
  const size_t bt = idx / DI_;
  const int t = (int)(bt % T_);
  const size_t brow = bt - t;
  float aq = cqb[d], ak = ckb[d], av = cvb[d];
#pragma unroll
  for (int j = 0; j < 4; ++j) {
    const int tt = t - 3 + j;
    if (tt < 0) continue;
    const size_t src = (brow + tt) * DI_ + d;
    aq += cqw[d * 4 + j] * qraw[src];
    ak += ckw[d * 4 + j] * kraw[src];
    av += cvw[d * 4 + j] * vraw[src];
  }
  qc[idx] = aq;
  vc[idx] = av;
  const float c0 = poly[0], c1 = poly[1];
  kphi[idx] = ak * (c0 + c1 * ak);
}

// ---------------------------------------------------------------------------
// Per-chunk gradient + omega-window + momentum scan. Grid = B*H blocks.
// ---------------------------------------------------------------------------
__global__ void chunk_grad_scan_kernel(const float* __restrict__ kphi,
                                       const float* __restrict__ vc,
                                       const float* __restrict__ gammag,
                                       const float* __restrict__ eta,
                                       const float* __restrict__ theta,
                                       const float* __restrict__ Mstate,
                                       float* __restrict__ Sstate,
                                       float* __restrict__ Sall,
                                       int chunk) {
  const int bh = blockIdx.x;
  const int b = bh >> 3, h = bh & 7;
  const int tid = threadIdx.x;
  __shared__ float sM[D_ * D_];
  __shared__ float sG[D_ * D_];
  __shared__ float ring_err[OMEGA_][D_];
  __shared__ float ring_kp[OMEGA_][D_];
  __shared__ float ring_g[OMEGA_];
  __shared__ float skp[D_];
  __shared__ float serr[D_];

  for (int i = tid; i < D_ * D_; i += 256) { sM[i] = Mstate[(size_t)bh * 4096 + i]; sG[i] = 0.0f; }
  float Sreg[16];
#pragma unroll
  for (int r = 0; r < 16; ++r) Sreg[r] = Sstate[(size_t)bh * 4096 + tid * 16 + r];
  __syncthreads();

  for (int tc = 0; tc < CS_; ++tc) {
    const int t = chunk * CS_ + tc;
    const size_t base = ((size_t)b * T_ + t) * DI_ + (size_t)h * D_;
    if (tid < D_) skp[tid] = kphi[base + tid];
    __syncthreads();
    if (tid < D_) {
      float p = 0.0f;
      for (int e = 0; e < D_; ++e) p += sM[tid * D_ + e] * skp[e];
      serr[tid] = p - vc[base + tid];
    }
    __syncthreads();
    const size_t gidx = ((size_t)b * T_ + t) * H_ + h;
    const float gmt = gammag[gidx];
    const float ett = eta[gidx];
    const float tht = theta[gidx];
    const int slot = tc & (OMEGA_ - 1);
    const bool haveold = (tc >= OMEGA_);
    const float gmold = haveold ? ring_g[slot] : 0.0f;
    const int gbase = tid * 16;
#pragma unroll
    for (int r = 0; r < 16; ++r) {
      const int fi = gbase + r;
      const int d = fi >> 6, e2 = fi & 63;
      float G = sG[fi] + 2.0f * gmt * serr[d] * skp[e2];
      if (haveold) G -= 2.0f * gmold * ring_err[slot][d] * ring_kp[slot][e2];
      sG[fi] = G;
      Sreg[r] = tht * Sreg[r] - ett * G;
      Sall[((size_t)tc * 16 + bh) * 4096 + fi] = Sreg[r];
    }
    __syncthreads();
    if (tid < D_) { ring_err[slot][tid] = serr[tid]; ring_kp[slot][tid] = skp[tid]; }
    if (tid == 0) ring_g[slot] = gmt;
    __syncthreads();
  }
#pragma unroll
  for (int r = 0; r < 16; ++r) Sstate[(size_t)bh * 4096 + tid * 16 + r] = Sreg[r];
}

// ---------------------------------------------------------------------------
// 64x64 matmul from bf16 LDS, both operands row-contiguous.
// Each wave computes tiles (wave) and (wave+8); both share the same ni, so
// the B fragment is loaded ONCE per kb and feeds two independent WMMAs.
// ---------------------------------------------------------------------------
__device__ __forceinline__ void mm64bf(const __bf16* Ar, const __bf16* Bt,
                                       int lane, int wave, v8f acc[2]) {
  const int offA = (lane < 16) ? 0 : 8;
  const int offB = (lane < 16) ? 0 : 8;
  const int mi0 = (wave >> 2) << 4;          // tile wave   -> rows mi0..mi0+15
  const int ni  = (wave & 3) << 4;           // tile wave+8 -> rows mi0+32..
  acc[0] = (v8f){};
  acc[1] = (v8f){};
#pragma unroll
  for (int kb = 0; kb < D_; kb += 32) {
    const __bf16* br = Bt + (ni + (lane & 15)) * D_;
    const v16bf bv = cat8(*(const v8bf*)(br + kb + offB),
                          *(const v8bf*)(br + kb + 16 + offB));
    const __bf16* ar0 = Ar + (mi0 + (lane & 15)) * D_;
    const v16bf av0 = cat8(*(const v8bf*)(ar0 + kb + offA),
                           *(const v8bf*)(ar0 + kb + 16 + offA));
    acc[0] = __builtin_amdgcn_wmma_f32_16x16x32_bf16(
        false, av0, false, bv, (short)0, acc[0], false, false);
    const __bf16* ar1 = Ar + (mi0 + 32 + (lane & 15)) * D_;
    const v16bf av1 = cat8(*(const v8bf*)(ar1 + kb + offA),
                           *(const v8bf*)(ar1 + kb + 16 + offA));
    acc[1] = __builtin_amdgcn_wmma_f32_16x16x32_bf16(
        false, av1, false, bv, (short)0, acc[1], false, false);
  }
}

// ---------------------------------------------------------------------------
// Polar-express (Newton-Schulz) on one 64x64 matrix per block.
// X tile is DMA'd into LDS by the Tensor Data Mover (one D#, TENSORcnt),
// bf16 LDS operands + transposed mirrors keep every fragment read contiguous.
// ---------------------------------------------------------------------------
__global__ void polar_kernel(const float* __restrict__ Sall, float* __restrict__ Spall) {
  __shared__ __align__(16) float  sXf[D_ * D_];     // f32 X (carried)
  __shared__ __align__(16) __bf16 sXb[D_ * D_];     // X   row-major
  __shared__ __align__(16) __bf16 sXbT[D_ * D_];    // X^T row-major
  __shared__ __align__(16) __bf16 sAb[D_ * D_];     // A   row-major
  __shared__ __align__(16) __bf16 sAbT[D_ * D_];    // A^T row-major
  __shared__ __align__(16) __bf16 sBmb[D_ * D_];    // Bm  row-major
  __shared__ float snorm;
  const int tid = threadIdx.x;
  const int lane = tid & 31, wave = tid >> 5;
  const size_t mat = (size_t)blockIdx.x * 4096;

#if __has_builtin(__builtin_amdgcn_tensor_load_to_lds)
  // TDM: DMA the 16KB X tile global->LDS. Issue from wave 0 only (TDM ignores
  // EXEC, so per-wave issue must be guarded at branch level).
  if (tid < 32) {
    const uint64_t ga = (uint64_t)(uintptr_t)(Sall + mat);
    const uint32_t lds = (uint32_t)(uintptr_t)(&sXf[0]);   // LDS aperture: addr[31:0]
    u32x4 g0;
    g0[0] = 1u;                                            // count=1, user D#
    g0[1] = lds;                                           // lds_addr
    g0[2] = (uint32_t)ga;                                  // global_addr[31:0]
    g0[3] = (uint32_t)((ga >> 32) & 0x1FFFFFFu) | (2u << 30); // addr[56:32] | type=2
    i32x8 g1;
    g1[0] = 0x00020000;                                    // data_size=4B, no flags
    g1[1] = (int)(4096u << 16);                            // tensor_dim0=4096 (lo16)
    g1[2] = (int)(1u << 16);                               // tensor_dim0 hi=0 | tensor_dim1=1
    g1[3] = (int)(4096u << 16);                            // tensor_dim1 hi=0 | tile_dim0=4096
    g1[4] = 1;                                             // tile_dim1=1, tile_dim2=0
    g1[5] = 4096;                                          // tensor_dim0_stride lo
    g1[6] = 0;                                             // stride0 hi | stride1 lo
    g1[7] = 0;
    i32x4 gz4 = {0, 0, 0, 0};
    i32x8 gz8 = {0, 0, 0, 0, 0, 0, 0, 0};
    __builtin_amdgcn_tensor_load_to_lds(g0, g1, gz4, gz4, gz8, 0);
    __builtin_amdgcn_s_wait_tensorcnt(0);
  }
  __syncthreads();
  asm volatile("" ::: "memory");
#else
  for (int i = tid; i < 4096; i += 256) sXf[i] = Sall[mat + i];
#endif

  if (tid == 0) snorm = 1e-12f;
  __syncthreads();
  float p = 0.0f;
  for (int i = tid; i < 4096; i += 256) p += sXf[i] * sXf[i];
  atomicAdd(&snorm, p);
  __syncthreads();
  const float inv = 1.0f / (sqrtf(snorm) * 1.01f + 1e-6f);
  for (int i = tid; i < 4096; i += 256) {
    const float v = sXf[i] * inv;
    sXf[i] = v;
    const __bf16 bv = f2bf(v);
    sXb[i] = bv;
    sXbT[(i & 63) * D_ + (i >> 6)] = bv;
  }
  __syncthreads();

  const float pea[3] = {8.156554524902461f, 4.042929935166739f, 3.8916678022926607f};
  const float peb[3] = {-22.48329292557795f, -2.808917465908714f, -2.772484153217685f};
  const float pec[3] = {15.878769915207462f, 0.5000178451051316f, 0.5060648178503393f};

  const int rb = (lane < 16) ? 0 : 8;
#pragma unroll 1
  for (int it = 0; it < 3; ++it) {
    v8f accA[2], accB[2];
    // 1) A = X X^T : B^T rows == X rows
    mm64bf(sXb, sXb, lane, wave, accA);
#pragma unroll
    for (int tt = 0; tt < 2; ++tt) {
      const int tile = wave + tt * 8;
      const int mi = (tile >> 2) << 4, ni = (tile & 3) << 4;
      const int nc = ni + (lane & 15);
#pragma unroll
      for (int r = 0; r < 8; ++r) {
        const int row = mi + rb + r;
        const __bf16 bv = f2bf(accA[tt][r]);
        sAb[row * D_ + nc]  = bv;
        sAbT[nc * D_ + row] = bv;
      }
    }
    __syncthreads();
    // 2) A2 = A A ; Bm = b*A + c*A2 (A tiles still live in accA)
    mm64bf(sAb, sAbT, lane, wave, accB);
#pragma unroll
    for (int tt = 0; tt < 2; ++tt) {
      const int tile = wave + tt * 8;
      const int mi = (tile >> 2) << 4, ni = (tile & 3) << 4;
      const int nc = ni + (lane & 15);
#pragma unroll
      for (int r = 0; r < 8; ++r) {
        const int row = mi + rb + r;
        sBmb[row * D_ + nc] = f2bf(peb[it] * accA[tt][r] + pec[it] * accB[tt][r]);
      }
    }
    __syncthreads();
    // 3) T = Bm X ; X = a*X + T
    mm64bf(sBmb, sXbT, lane, wave, accB);
    __syncthreads();                    // all waves done reading sXbT/sBmb
#pragma unroll
    for (int tt = 0; tt < 2; ++tt) {
      const int tile = wave + tt * 8;
      const int mi = (tile >> 2) << 4, ni = (tile & 3) << 4;
      const int nc = ni + (lane & 15);
#pragma unroll
      for (int r = 0; r < 8; ++r) {
        const int row = mi + rb + r;
        const int pos = row * D_ + nc;
        const float xn = pea[it] * sXf[pos] + accB[tt][r];
        sXf[pos] = xn;
        const __bf16 bv = f2bf(xn);
        sXb[pos] = bv;
        sXbT[nc * D_ + row] = bv;
      }
    }
    __syncthreads();
  }
  for (int i = tid; i < 4096; i += 256) Spall[mat + i] = sXf[i];
}

// ---------------------------------------------------------------------------
// Per-chunk memory scan M_t = alpha_t*M_{t-1} + S'_t and readout y = M q.
// ---------------------------------------------------------------------------
__global__ void chunk_mem_out_kernel(const float* __restrict__ Spall,
                                     const float* __restrict__ qc,
                                     const float* __restrict__ alpha,
                                     float* __restrict__ Mstate,
                                     float* __restrict__ ybuf,
                                     int chunk) {
  const int bh = blockIdx.x;
  const int b = bh >> 3, h = bh & 7;
  const int tid = threadIdx.x;
  __shared__ float sq[D_];
  __shared__ float sy[D_];
  float Mreg[16];
#pragma unroll
  for (int r = 0; r < 16; ++r) Mreg[r] = Mstate[(size_t)bh * 4096 + tid * 16 + r];
  const int d0 = (tid * 16) >> 6;
  const int e0 = (tid * 16) & 63;

  for (int tc = 0; tc < CS_; ++tc) {
    const int t = chunk * CS_ + tc;
    const float* Sp = Spall + ((size_t)tc * 16 + bh) * 4096;
    const float al = alpha[((size_t)b * T_ + t) * H_ + h];
#pragma unroll
    for (int r = 0; r < 16; ++r) Mreg[r] = al * Mreg[r] + Sp[tid * 16 + r];
    if (tid < D_) {
      sq[tid] = qc[((size_t)b * T_ + t) * DI_ + (size_t)h * D_ + tid];
      sy[tid] = 0.0f;
    }
    __syncthreads();
    float part = 0.0f;
#pragma unroll
    for (int r = 0; r < 16; ++r) part += Mreg[r] * sq[e0 + r];
    atomicAdd(&sy[d0], part);
    __syncthreads();
    if (tid < D_) ybuf[((size_t)b * T_ + t) * DI_ + (size_t)h * D_ + tid] = sy[tid];
    __syncthreads();
  }
#pragma unroll
  for (int r = 0; r < 16; ++r) Mstate[(size_t)bh * 4096 + tid * 16 + r] = Mreg[r];
}

// ---------------------------------------------------------------------------
// RMS-norm over D, (1+ln_gamma) scale, retrieve gate; emits bf16 directly.
// ---------------------------------------------------------------------------
__global__ void norm_gate_kernel(const float* __restrict__ ybuf,
                                 const float* __restrict__ ln_gamma,
                                 const float* __restrict__ rgate,
                                 __bf16* __restrict__ ygb) {
  __shared__ float ssum[4];
  const int grp = threadIdx.x >> 6;
  const int d = threadIdx.x & 63;
  const size_t idx = (size_t)blockIdx.x * 4 + grp;    // bt*H + h
  const int h = (int)(idx % H_);
  const float v = ybuf[idx * D_ + d];
  if (d == 0) ssum[grp] = 0.0f;
  __syncthreads();
  atomicAdd(&ssum[grp], v * v);
  __syncthreads();
  const float ms = ssum[grp] * (1.0f / 64.0f);
  float nv = v * rsqrtf(ms + 1e-6f);
  nv *= (1.0f + ln_gamma[h * D_ + d]);
  nv *= rgate[idx];
  ygb[idx * D_ + d] = f2bf(nv);
}

// ---------------------------------------------------------------------------
extern "C" void kernel_launch(void* const* d_in, const int* in_sizes, int n_in,
                              void* d_out, int out_size, void* d_ws, size_t ws_size,
                              hipStream_t stream) {
  (void)in_sizes; (void)n_in; (void)out_size; (void)ws_size;
  const float* x      = (const float*)d_in[0];
  const float* Wq     = (const float*)d_in[1];
  const float* Wk     = (const float*)d_in[2];
  const float* Wv     = (const float*)d_in[3];
  const float* Wproj  = (const float*)d_in[4];
  const float* cq_w   = (const float*)d_in[5];
  const float* cq_b   = (const float*)d_in[6];
  const float* ck_w   = (const float*)d_in[7];
  const float* ck_b   = (const float*)d_in[8];
  const float* cv_w   = (const float*)d_in[9];
  const float* cv_b   = (const float*)d_in[10];
  const float* ga_w   = (const float*)d_in[11];
  const float* ga_b   = (const float*)d_in[12];
  const float* ge_w   = (const float*)d_in[13];
  const float* ge_b   = (const float*)d_in[14];
  const float* gt_w   = (const float*)d_in[15];
  const float* gt_b   = (const float*)d_in[16];
  const float* gg_w   = (const float*)d_in[17];
  const float* gg_b   = (const float*)d_in[18];
  const float* poly   = (const float*)d_in[19];
  const float* ln_g   = (const float*)d_in[20];
  const float* rg_w   = (const float*)d_in[21];
  float* out = (float*)d_out;

  float* ws = (float*)d_ws;
  size_t o = 0;
  float* qraw   = ws + o; o += (size_t)BT_ * DI_;
  float* kraw   = ws + o; o += (size_t)BT_ * DI_;
  float* vraw   = ws + o; o += (size_t)BT_ * DI_;
  float* qc     = ws + o; o += (size_t)BT_ * DI_;
  float* kphi   = ws + o; o += (size_t)BT_ * DI_;
  float* vc     = ws + o; o += (size_t)BT_ * DI_;
  float* alpha  = ws + o; o += (size_t)BT_ * H_;
  float* eta    = ws + o; o += (size_t)BT_ * H_;
  float* theta  = ws + o; o += (size_t)BT_ * H_;
  float* gammag = ws + o; o += (size_t)BT_ * H_;
  float* rgate  = ws + o; o += (size_t)BT_ * H_;
  float* Mstate = ws + o; o += (size_t)B_ * H_ * D_ * D_;
  float* Sstate = ws + o; o += (size_t)B_ * H_ * D_ * D_;
  float* Sall   = ws + o; o += (size_t)B_ * CS_ * H_ * D_ * D_;
  float* Spall  = ws + o; o += (size_t)B_ * CS_ * H_ * D_ * D_;
  float* ybuf   = ws + o; o += (size_t)BT_ * DI_;
  __bf16* xb  = (__bf16*)(ws + o); o += (size_t)BT_ * C_ / 2;
  __bf16* Wqb = (__bf16*)(ws + o); o += (size_t)DI_ * C_ / 2;
  __bf16* Wkb = (__bf16*)(ws + o); o += (size_t)DI_ * C_ / 2;
  __bf16* Wvb = (__bf16*)(ws + o); o += (size_t)DI_ * C_ / 2;
  __bf16* Wpb = (__bf16*)(ws + o); o += (size_t)C_ * DI_ / 2;
  __bf16* ygb = (__bf16*)(ws + o); o += (size_t)BT_ * DI_ / 2;

  cvt_bf16_kernel<<<(BT_ * C_) / 256, 256, 0, stream>>>(x, xb, BT_ * C_);
  cvt_bf16_kernel<<<(DI_ * C_) / 256, 256, 0, stream>>>(Wq, Wqb, DI_ * C_);
  cvt_bf16_kernel<<<(DI_ * C_) / 256, 256, 0, stream>>>(Wk, Wkb, DI_ * C_);
  cvt_bf16_kernel<<<(DI_ * C_) / 256, 256, 0, stream>>>(Wv, Wvb, DI_ * C_);
  cvt_bf16_kernel<<<(C_ * DI_) / 256, 256, 0, stream>>>(Wproj, Wpb, C_ * DI_);

  gemm_bf16_kernel<<<128, 256, 0, stream>>>(xb, Wqb, qraw, BT_, DI_, C_);
  gemm_bf16_kernel<<<128, 256, 0, stream>>>(xb, Wkb, kraw, BT_, DI_, C_);
  gemm_bf16_kernel<<<128, 256, 0, stream>>>(xb, Wvb, vraw, BT_, DI_, C_);

  gates_kernel<<<BT_, 256, 0, stream>>>(x, ga_w, ga_b, ge_w, ge_b, gt_w, gt_b,
                                        gg_w, gg_b, rg_w,
                                        alpha, eta, theta, gammag, rgate);

  conv_poly_kernel<<<(BT_ * DI_) / 256, 256, 0, stream>>>(
      qraw, kraw, vraw, cq_w, cq_b, ck_w, ck_b, cv_w, cv_b, poly, qc, kphi, vc);

  (void)hipMemsetAsync(Mstate, 0, (size_t)2 * B_ * H_ * D_ * D_ * sizeof(float), stream); // M & S

  for (int c = 0; c < NC_; ++c) {
    chunk_grad_scan_kernel<<<B_ * H_, 256, 0, stream>>>(
        kphi, vc, gammag, eta, theta, Mstate, Sstate, Sall, c);
    polar_kernel<<<CS_ * B_ * H_, 256, 0, stream>>>(Sall, Spall);
    chunk_mem_out_kernel<<<B_ * H_, 256, 0, stream>>>(
        Spall, qc, alpha, Mstate, ybuf, c);
  }

  norm_gate_kernel<<<(BT_ * H_) / 4, 256, 0, stream>>>(ybuf, ln_g, rgate, ygb);

  gemm_bf16_kernel<<<256, 256, 0, stream>>>(ygb, Wpb, out, BT_, C_, DI_);
}
